// NCCNet_8134668059174
// MI455X (gfx1250) — matrix operand order repaired and checked by reading, not compiled, not run on hardware
//
#include <hip/hip_runtime.h>
#include <hip/hip_fp16.h>

typedef __attribute__((ext_vector_type(16))) _Float16 v16h;
typedef __attribute__((ext_vector_type(8)))  float    v8f;
typedef __attribute__((ext_vector_type(4)))  unsigned int tdm_u32x4;
typedef __attribute__((ext_vector_type(8)))  int          tdm_i32x8;
typedef __attribute__((ext_vector_type(4)))  int          tdm_i32x4;

// ---------------- problem dimensions ----------------
constexpr int B_   = 64;
constexpr int C1O  = 20;           // conv1 out channels
constexpr int H1p  = 78, W1p = 28; // after conv1+pool
constexpr int C2O  = 25;           // conv2 out channels (feature C)
constexpr int FH   = 37, FW = 12;  // feature map H,W
constexpr int NPIX = FH * FW;      // 444
constexpr int YH   = FH + 4;       // 41 rows of Y patch starts
constexpr int NCH  = 1500;         // C*5*W correlation channels
// padded (TDM-friendly) extents
constexpr int NCHP  = 1504;        // 47*32
constexpr int NPIXP = 448;         // 28*16
constexpr int KFC   = 2125;
constexpr int KFCP  = 2144;        // 67*32
constexpr int NFC   = 500;
constexpr int NFCP  = 512;

// ---------------- WMMA fragment index helpers (wave32) ----------------
__device__ __forceinline__ int a_kidx(int e, int hi) {
    return (e < 8) ? (e + 8 * hi) : (16 + (e - 8) + 8 * hi);
}
__device__ __forceinline__ int b_kidx(int e, int hi) {
    return (e < 8) ? (e + 8 * hi) : (16 + (e - 8) + 8 * hi);
}
// C/D f32 16x16: VGPR r -> row = 8*hi + r, col = lane&15

// ---------------- TDM 2D f16 tile load: global -> LDS ----------------
// Loads tile_y rows of tile_x contiguous f16 elements (row stride = stride_x
// elements) starting at gaddr into LDS at lds_addr, packed contiguously.
__device__ __forceinline__ void tdm_load_2d_f16(unsigned lds_addr, const void* gaddr,
                                                unsigned tile_x, unsigned tile_y,
                                                unsigned tensor_x, unsigned tensor_y,
                                                unsigned stride_x) {
    unsigned long long ga = (unsigned long long)(size_t)gaddr;
    tdm_u32x4 g0;
    g0[0] = 1u;                                              // count=1, user mode
    g0[1] = lds_addr;                                        // LDS byte address
    g0[2] = (unsigned)(ga & 0xFFFFFFFFu);                    // global_addr[31:0]
    g0[3] = (unsigned)((ga >> 32) & 0x1FFFFFFu) | (2u << 30);// global_addr[56:32] | type=2
    tdm_i32x8 g1;
    g1[0] = (int)(1u << 16);                                 // data_size=1 (2B), mask=0
    g1[1] = (int)((tensor_x & 0xFFFFu) << 16);               // tensor_dim0[15:0]
    g1[2] = (int)(((tensor_x >> 16) & 0xFFFFu) | ((tensor_y & 0xFFFFu) << 16));
    g1[3] = (int)(((tensor_y >> 16) & 0xFFFFu) | ((tile_x & 0xFFFFu) << 16));
    g1[4] = (int)(tile_y & 0xFFFFu);                         // tile_dim1; tile_dim2=0
    g1[5] = (int)stride_x;                                   // tensor_dim0_stride[31:0]
    g1[6] = 0;                                               // stride hi | dim1_stride lo
    g1[7] = 0;
    tdm_i32x4 gz4 = {0, 0, 0, 0};
    tdm_i32x8 gz8 = {0, 0, 0, 0, 0, 0, 0, 0};
    __builtin_amdgcn_tensor_load_to_lds(g0, g1, gz4, gz4, gz8, 0);
}

// ---------------- tower stage 1: conv(3->20,5x5)+relu+maxpool2 ----------------
__global__ void conv1_pool_kernel(const float* __restrict__ x,
                                  const float* __restrict__ w,
                                  const float* __restrict__ bias,
                                  float* __restrict__ out) {
    int idx = blockIdx.x * blockDim.x + threadIdx.x;
    const int total = B_ * C1O * H1p * W1p;
    if (idx >= total) return;
    int pw = idx % W1p; int t = idx / W1p;
    int ph = t % H1p;   t /= H1p;
    int oc = t % C1O;   int b = t / C1O;
    float mx = -3.0e38f;
    for (int r = 0; r < 2; ++r) {
        for (int s = 0; s < 2; ++s) {
            int oh = 2 * ph + r, ow = 2 * pw + s;
            float acc = bias[oc];
            for (int ic = 0; ic < 3; ++ic)
                for (int ki = 0; ki < 5; ++ki)
                    for (int kj = 0; kj < 5; ++kj)
                        acc += x[((size_t)(b * 3 + ic) * 160 + (oh + ki)) * 60 + (ow + kj)]
                             * w[((oc * 3 + ic) * 5 + ki) * 5 + kj];
            acc = acc > 0.f ? acc : 0.f;
            mx = fmaxf(mx, acc);
        }
    }
    out[idx] = mx;
}

// ---------------- tower stage 2: conv(20->25,5x5)+relu+maxpool2 ----------------
__global__ void conv2_pool_kernel(const float* __restrict__ in,
                                  const float* __restrict__ w,
                                  const float* __restrict__ bias,
                                  float* __restrict__ out) {
    int idx = blockIdx.x * blockDim.x + threadIdx.x;
    const int total = B_ * C2O * FH * FW;
    if (idx >= total) return;
    int pw = idx % FW; int t = idx / FW;
    int ph = t % FH;   t /= FH;
    int oc = t % C2O;  int b = t / C2O;
    float mx = -3.0e38f;
    for (int r = 0; r < 2; ++r) {
        for (int s = 0; s < 2; ++s) {
            int oh = 2 * ph + r, ow = 2 * pw + s;
            float acc = bias[oc];
            for (int ic = 0; ic < C1O; ++ic)
                for (int ki = 0; ki < 5; ++ki)
                    for (int kj = 0; kj < 5; ++kj)
                        acc += in[((size_t)(b * C1O + ic) * H1p + (oh + ki)) * W1p + (ow + kj)]
                             * w[((oc * C1O + ic) * 5 + ki) * 5 + kj];
            acc = acc > 0.f ? acc : 0.f;
            mx = fmaxf(mx, acc);
        }
    }
    out[idx] = mx;
}

// ---------------- patch stats (mean, 1/(std+eps)) ----------------
__global__ void stats_x_kernel(const float* __restrict__ f,
                               float* __restrict__ mu, float* __restrict__ inv) {
    int idx = blockIdx.x * blockDim.x + threadIdx.x;
    const int total = B_ * C2O * NPIX;
    if (idx >= total) return;
    int j = idx % FW; int t = idx / FW;
    int i = t % FH;   int bc = t / FH;
    const float* fp = f + (size_t)bc * NPIX;
    float s1 = 0.f, s2 = 0.f;
    for (int di = 0; di < 5; ++di)
        for (int dj = 0; dj < 5; ++dj) {
            int ry = i + di - 2, rx = j + dj - 2;
            float v = 0.f;
            if (ry >= 0 && ry < FH && rx >= 0 && rx < FW) v = fp[ry * FW + rx];
            s1 += v; s2 += v * v;
        }
    float m = s1 * (1.f / 25.f);
    float var = s2 * (1.f / 25.f) - m * m;
    var = var > 0.f ? var : 0.f;
    mu[idx]  = m;
    inv[idx] = 1.f / (sqrtf(var) + 0.01f);
}

__global__ void stats_y_kernel(const float* __restrict__ f,
                               float* __restrict__ mu, float* __restrict__ inv) {
    int idx = blockIdx.x * blockDim.x + threadIdx.x;
    const int total = B_ * C2O * YH * FW;
    if (idx >= total) return;
    int x = idx % FW; int t = idx / FW;
    int iy = t % YH;  int bc = t / YH;
    const float* fp = f + (size_t)bc * NPIX;
    float s1 = 0.f, s2 = 0.f;
    for (int di = 0; di < 5; ++di)
        for (int dj = 0; dj < 5; ++dj) {
            int ry = iy + di - 4, rx = x + dj - 2;
            float v = 0.f;
            if (ry >= 0 && ry < FH && rx >= 0 && rx < FW) v = fp[ry * FW + rx];
            s1 += v; s2 += v * v;
        }
    float m = s1 * (1.f / 25.f);
    float var = s2 * (1.f / 25.f) - m * m;
    var = var > 0.f ? var : 0.f;
    mu[idx]  = m;
    inv[idx] = 1.f / (sqrtf(var) + 0.01f);
}

// ---------------- prep: zero fill (u32 granularity) ----------------
__global__ void zero_fill_kernel(unsigned int* __restrict__ p, long long n) {
    long long i = (long long)blockIdx.x * blockDim.x + threadIdx.x;
    if (i < n) p[i] = 0u;
}

// ---------------- prep: f32 matrix -> zero-padded f16 matrix ----------------
__global__ void cvt_pad_f16_kernel(const float* __restrict__ src, _Float16* __restrict__ dst,
                                   int rows, int cols, int rows_p, int cols_p) {
    long long i = (long long)blockIdx.x * blockDim.x + threadIdx.x;
    long long tot = (long long)rows_p * cols_p;
    if (i >= tot) return;
    int c = (int)(i % cols_p), r = (int)(i / cols_p);
    float v = (r < rows && c < cols) ? src[(size_t)r * cols + c] : 0.f;
    dst[i] = (_Float16)v;
}

// ---------------- normxcorr via WMMA (LDS-staged windows) ----------------
__global__ void normxcorr_wmma_kernel(const float* __restrict__ fx,
                                      const float* __restrict__ fy,
                                      const float* __restrict__ muX,
                                      const float* __restrict__ invX,
                                      const float* __restrict__ muY,
                                      const float* __restrict__ invY,
                                      _Float16* __restrict__ rbuf) {
    __shared__ float lx[7][20];   // X rows i-2..i+4, cols -2..17 (zero padded)
    __shared__ float ly[10][16];  // Y rows i-4..i+5, cols -2..13 (zero padded)

    int blk = blockIdx.x;              // b*C2O*FH + c*FH + i
    int i = blk % FH; int t = blk / FH;
    int c = t % C2O;  int b = t / C2O;
    int lane = threadIdx.x & 31;
    int l15 = lane & 15, hi = lane >> 4;

    const size_t bc = (size_t)(b * C2O + c);
    const float* fxp = fx + bc * NPIX;
    const float* fyp = fy + bc * NPIX;

    for (int tt = lane; tt < 7 * 20; tt += 32) {
        int rr = tt / 20, cc = tt % 20;
        int ry = i + rr - 2, rx = cc - 2;
        float v = 0.f;
        if (ry >= 0 && ry < FH && rx >= 0 && rx < FW) v = fxp[ry * FW + rx];
        lx[rr][cc] = v;
    }
    for (int tt = lane; tt < 10 * 16; tt += 32) {
        int rr = tt / 16, cc = tt % 16;
        int ry = i + rr - 4, rx = cc - 2;
        float v = 0.f;
        if (ry >= 0 && ry < FH && rx >= 0 && rx < FW) v = fyp[ry * FW + rx];
        ly[rr][cc] = v;
    }
    __syncthreads();

    v16h bfrag;
#pragma unroll
    for (int e = 0; e < 16; ++e) {
        int p = b_kidx(e, hi);       // 0..31; p>=25 zeroed on A side
        int di = p / 5, dj = p % 5;
        di = di > 6 ? 6 : di;
        bfrag[e] = (_Float16)lx[di][l15 + dj];
    }

    float mX = 0.f, iX = 0.f;
    if (l15 < FW) {
        size_t sx = bc * NPIX + i * FW + l15;
        mX = muX[sx]; iX = invX[sx];
    }

#pragma unroll
    for (int mt = 0; mt < 4; ++mt) {
        int m = mt * 16 + l15;           // row = d*12 + x
        int d = m / FW, x = m % FW;
        v16h afrag;
#pragma unroll
        for (int e = 0; e < 16; ++e) {
            int p = a_kidx(e, hi);
            int di = p / 5, dj = p % 5;
            float v = (p < 25) ? ly[d + di][x + dj] : 0.f;
            afrag[e] = (_Float16)v;
        }
        v8f acc = {};
        acc = __builtin_amdgcn_wmma_f32_16x16x32_f16(false, afrag, false, bfrag,
                                                     (short)0, acc, false, false);
#pragma unroll
        for (int r = 0; r < 8; ++r) {
            int mm = mt * 16 + hi * 8 + r;
            int n  = l15;                 // j
            if (mm < 60 && n < FW) {
                int dd = mm / FW, xx = mm % FW;
                size_t sy = bc * (YH * FW) + (size_t)(i + dd) * FW + xx;
                float mY = muY[sy], iY = invY[sy];
                float res = (acc[r] - 25.f * mX * mY) * iX * iY * (1.f / 25.f);
                res = res > 0.f ? res : 0.f;
                int ch = c * 60 + dd * FW + xx;
                rbuf[((size_t)b * NCHP + ch) * NPIXP + i * FW + n] = (_Float16)res;
            }
        }
    }
}

// ---------------- dr1: per-batch GEMM W[25,1500] x r[1500,444] (TDM-staged) ----------------
__global__ void dr1_wmma_kernel(const _Float16* __restrict__ rbuf,
                                const _Float16* __restrict__ w16,  // [32][1504]
                                const float* __restrict__ bias,
                                float* __restrict__ out) {
    __shared__ _Float16 lds_a[16][32];   // A tile, m-major
    __shared__ _Float16 lds_b[32][16];   // B tile, k-major

    int nt = blockIdx.x;  // 0..27
    int mt = blockIdx.y;  // 0..1
    int b  = blockIdx.z;
    int lane = threadIdx.x & 31;
    int l15 = lane & 15, hi = lane >> 4;
    const _Float16* rb = rbuf + (size_t)b * NCHP * NPIXP;
    const int m0 = mt * 16, n0 = nt * 16;
    const unsigned lds_a_addr = (unsigned)(size_t)&lds_a[0][0];
    const unsigned lds_b_addr = (unsigned)(size_t)&lds_b[0][0];

    v8f acc = {};
    for (int kt = 0; kt < 47; ++kt) {
        int k0 = kt * 32;
        // TDM: A tile = w16[m0..m0+15][k0..k0+31] -> lds_a[16][32]
        tdm_load_2d_f16(lds_a_addr, &w16[(size_t)m0 * NCHP + k0],
                        32, 16, NCHP, 32, NCHP);
        // TDM: B tile = rb[k0..k0+31][n0..n0+15] -> lds_b[32][16]
        tdm_load_2d_f16(lds_b_addr, &rb[(size_t)k0 * NPIXP + n0],
                        16, 32, NPIXP, NCHP, NPIXP);
        if (k0 + 32 < NCH)
            __builtin_prefetch(&rb[(size_t)(k0 + 32 + lane) * NPIXP + n0], 0, 1);
        __builtin_amdgcn_s_wait_tensorcnt(0);
        __syncthreads();

        v16h a, bf;
#pragma unroll
        for (int e = 0; e < 16; ++e) a[e]  = lds_a[l15][a_kidx(e, hi)];
#pragma unroll
        for (int e = 0; e < 16; ++e) bf[e] = lds_b[b_kidx(e, hi)][l15];
        acc = __builtin_amdgcn_wmma_f32_16x16x32_f16(false, a, false, bf,
                                                     (short)0, acc, false, false);
        __syncthreads();
    }
#pragma unroll
    for (int r = 0; r < 8; ++r) {
        int mm = m0 + hi * 8 + r;
        int nn = n0 + l15;
        if (mm < C2O && nn < NPIX) {
            float v = acc[r] + bias[mm];
            out[((size_t)b * C2O + mm) * NPIX + nn] = v > 0.f ? v : 0.f;
        }
    }
}

// ---------------- dr2: conv(25->25,3x3) + maxpool2 ----------------
__global__ void dr2_pool_kernel(const float* __restrict__ in,
                                const float* __restrict__ w,
                                const float* __restrict__ bias,
                                float* __restrict__ out) {
    int idx = blockIdx.x * blockDim.x + threadIdx.x;
    const int total = B_ * C2O * 17 * 5;
    if (idx >= total) return;
    int pw = idx % 5; int t = idx / 5;
    int ph = t % 17;  t /= 17;
    int oc = t % C2O; int b = t / C2O;
    float mx = -3.0e38f;
    for (int r = 0; r < 2; ++r) {
        for (int s = 0; s < 2; ++s) {
            int oh = 2 * ph + r, ow = 2 * pw + s;
            float acc = bias[oc];
            for (int ic = 0; ic < C2O; ++ic)
                for (int ki = 0; ki < 3; ++ki)
                    for (int kj = 0; kj < 3; ++kj)
                        acc += in[((size_t)b * C2O + ic) * NPIX + (oh + ki) * FW + (ow + kj)]
                             * w[((oc * C2O + ic) * 3 + ki) * 3 + kj];
            mx = fmaxf(mx, acc);
        }
    }
    out[idx] = mx;
}

// ---------------- fc1: X[64,2125] x W^T + bias (TDM-staged WMMA) ----------------
__global__ void fc1_wmma_kernel(const _Float16* __restrict__ x16,  // [64][2144]
                                const _Float16* __restrict__ w16,  // [512][2144]
                                const float* __restrict__ bias,
                                float* __restrict__ out) {
    __shared__ _Float16 lds_a[16][32];   // A tile, m-major
    __shared__ _Float16 lds_bt[16][32];  // B tile, n-major (transposed staging)

    int nt = blockIdx.x;  // 0..31
    int mt = blockIdx.y;  // 0..3
    int lane = threadIdx.x & 31;
    int l15 = lane & 15, hi = lane >> 4;
    const int m0 = mt * 16, n0 = nt * 16;
    const unsigned lds_a_addr = (unsigned)(size_t)&lds_a[0][0];
    const unsigned lds_b_addr = (unsigned)(size_t)&lds_bt[0][0];

    v8f acc = {};
    for (int kt = 0; kt < 67; ++kt) {
        int k0 = kt * 32;
        // A tile = x16[m0..m0+15][k0..k0+31] -> lds_a[16][32]
        tdm_load_2d_f16(lds_a_addr, &x16[(size_t)m0 * KFCP + k0],
                        32, 16, KFCP, 64, KFCP);
        // B tile = w16[n0..n0+15][k0..k0+31] -> lds_bt[16][32]
        tdm_load_2d_f16(lds_b_addr, &w16[(size_t)n0 * KFCP + k0],
                        32, 16, KFCP, NFCP, KFCP);
        if (k0 + 32 < KFCP)
            __builtin_prefetch(&w16[(size_t)(n0 + l15) * KFCP + k0 + 32], 0, 1);
        __builtin_amdgcn_s_wait_tensorcnt(0);
        __syncthreads();

        v16h a, bf;
#pragma unroll
        for (int e = 0; e < 16; ++e) a[e]  = lds_a[l15][a_kidx(e, hi)];
#pragma unroll
        for (int e = 0; e < 16; ++e) bf[e] = lds_bt[l15][b_kidx(e, hi)];
        acc = __builtin_amdgcn_wmma_f32_16x16x32_f16(false, a, false, bf,
                                                     (short)0, acc, false, false);
        __syncthreads();
    }
#pragma unroll
    for (int r = 0; r < 8; ++r) {
        int mm = m0 + hi * 8 + r;
        int nn = n0 + l15;
        if (mm < 64 && nn < NFC)
            out[(size_t)mm * NFC + nn] = acc[r] + bias[nn];
    }
}

// ---------------- fc2: [64,500] -> [64,2] ----------------
__global__ void fc2_kernel(const float* __restrict__ xin,
                           const float* __restrict__ w,
                           const float* __restrict__ bias,
                           float* __restrict__ out) {
    int t = blockIdx.x * blockDim.x + threadIdx.x;
    if (t >= 128) return;
    int o = t & 1, b = t >> 1;
    float acc = bias[o];
    for (int k = 0; k < 500; ++k)
        acc += xin[b * 500 + k] * w[o * 500 + k];
    out[b * 2 + o] = acc;
}

// ---------------- launcher ----------------
static inline size_t alignup(size_t v) { return (v + 255) & ~(size_t)255; }

extern "C" void kernel_launch(void* const* d_in, const int* in_sizes, int n_in,
                              void* d_out, int out_size, void* d_ws, size_t ws_size,
                              hipStream_t stream) {
    const float* x1      = (const float*)d_in[0];
    const float* x2      = (const float*)d_in[1];
    const float* conv1_w = (const float*)d_in[2];
    const float* conv1_b = (const float*)d_in[3];
    const float* conv2_w = (const float*)d_in[4];
    const float* conv2_b = (const float*)d_in[5];
    const float* dr1_w   = (const float*)d_in[6];
    const float* dr1_b   = (const float*)d_in[7];
    const float* dr2_w   = (const float*)d_in[8];
    const float* dr2_b   = (const float*)d_in[9];
    const float* fc1_w   = (const float*)d_in[10];
    const float* fc1_b   = (const float*)d_in[11];
    const float* fc2_w   = (const float*)d_in[12];
    const float* fc2_b   = (const float*)d_in[13];

    char* ws = (char*)d_ws;
    size_t off = 0;
    const size_t S_p1    = (size_t)B_ * C1O * H1p * W1p * sizeof(float);
    const size_t S_feat  = (size_t)B_ * C2O * NPIX * sizeof(float);
    const size_t S_sy    = (size_t)B_ * C2O * YH * FW * sizeof(float);
    const size_t S_r16   = (size_t)B_ * NCHP * NPIXP * sizeof(_Float16); // 86.2 MB
    const size_t S_w1_16 = (size_t)32 * NCHP * sizeof(_Float16);
    const size_t S_x16   = (size_t)64 * KFCP * sizeof(_Float16);
    const size_t S_wf16  = (size_t)NFCP * KFCP * sizeof(_Float16);
    const size_t S_pool2 = (size_t)B_ * C2O * 17 * 5 * sizeof(float);
    const size_t S_fc1   = (size_t)64 * 500 * sizeof(float);

    float* fx_p1 = (float*)(ws + off); off += alignup(S_p1);
    float* fy_p1 = (float*)(ws + off); off += alignup(S_p1);
    float* fx    = (float*)(ws + off); off += alignup(S_feat);
    float* fy    = (float*)(ws + off); off += alignup(S_feat);
    float* muX   = (float*)(ws + off); off += alignup(S_feat);
    float* invX  = (float*)(ws + off); off += alignup(S_feat);
    float* muY   = (float*)(ws + off); off += alignup(S_sy);
    float* invY  = (float*)(ws + off); off += alignup(S_sy);
    _Float16* rbuf  = (_Float16*)(ws + off); off += alignup(S_r16);
    _Float16* w1_16 = (_Float16*)(ws + off); off += alignup(S_w1_16);
    _Float16* x16   = (_Float16*)(ws + off); off += alignup(S_x16);
    _Float16* wf16  = (_Float16*)(ws + off); off += alignup(S_wf16);
    float* dr1o  = (float*)(ws + off); off += alignup(S_feat);
    float* pool2 = (float*)(ws + off); off += alignup(S_pool2);
    float* fc1o  = (float*)(ws + off); off += alignup(S_fc1);
    (void)ws_size; (void)in_sizes; (void)n_in; (void)out_size;

    const int T = 256;
    // towers
    {
        int tot = B_ * C1O * H1p * W1p;
        conv1_pool_kernel<<<(tot + T - 1) / T, T, 0, stream>>>(x1, conv1_w, conv1_b, fx_p1);
        conv1_pool_kernel<<<(tot + T - 1) / T, T, 0, stream>>>(x2, conv1_w, conv1_b, fy_p1);
    }
    {
        int tot = B_ * C2O * FH * FW;
        conv2_pool_kernel<<<(tot + T - 1) / T, T, 0, stream>>>(fx_p1, conv2_w, conv2_b, fx);
        conv2_pool_kernel<<<(tot + T - 1) / T, T, 0, stream>>>(fy_p1, conv2_w, conv2_b, fy);
    }
    // patch stats
    {
        int totx = B_ * C2O * NPIX;
        stats_x_kernel<<<(totx + T - 1) / T, T, 0, stream>>>(fx, muX, invX);
        int toty = B_ * C2O * YH * FW;
        stats_y_kernel<<<(toty + T - 1) / T, T, 0, stream>>>(fy, muY, invY);
    }
    // prep: zero padded rbuf; convert weights to padded f16
    {
        long long n32 = (long long)(S_r16 / 4);
        zero_fill_kernel<<<(unsigned)((n32 + T - 1) / T), T, 0, stream>>>((unsigned int*)rbuf, n32);
        long long tw1 = (long long)32 * NCHP;
        cvt_pad_f16_kernel<<<(unsigned)((tw1 + T - 1) / T), T, 0, stream>>>(dr1_w, w1_16, C2O, NCH, 32, NCHP);
        long long twf = (long long)NFCP * KFCP;
        cvt_pad_f16_kernel<<<(unsigned)((twf + T - 1) / T), T, 0, stream>>>(fc1_w, wf16, NFC, KFC, NFCP, KFCP);
    }
    // normxcorr (WMMA) -> rbuf f16 [B,1504,448] (padded)
    normxcorr_wmma_kernel<<<B_ * C2O * FH, 32, 0, stream>>>(fx, fy, muX, invX, muY, invY, rbuf);
    // dr1 (WMMA + TDM staging)
    dr1_wmma_kernel<<<dim3(28, 2, B_), 32, 0, stream>>>(rbuf, w1_16, dr1_b, dr1o);
    // dr2 + pool
    {
        int tot = B_ * C2O * 17 * 5;
        dr2_pool_kernel<<<(tot + T - 1) / T, T, 0, stream>>>(dr1o, dr2_w, dr2_b, pool2);
    }
    // convert pooled activations to padded f16 for fc1
    {
        long long tx = (long long)64 * KFCP;
        cvt_pad_f16_kernel<<<(unsigned)((tx + T - 1) / T), T, 0, stream>>>(pool2, x16, 64, KFC, 64, KFCP);
    }
    // fc1 (WMMA + TDM staging), fc2
    fc1_wmma_kernel<<<dim3(32, 4, 1), 32, 0, stream>>>(x16, wf16, fc1_b, fc1o);
    fc2_kernel<<<1, 128, 0, stream>>>(fc1o, fc2_w, fc2_b, (float*)d_out);
}
